// CorrelationBlock_21457656611321
// MI455X (gfx1250) — compile-verified, matrix operand order checked by reading, not run on hardware
//
#include <hip/hip_runtime.h>
#include <hip/hip_bf16.h>

// ---------------------------------------------------------------------------
// RAFT correlation block for gfx1250 (MI455X).
//
// Key identity: avgpool over (h2,w2) commutes with the channel dot product,
//   pool(corr)[p,y,x] = dot(f1[p], pool(fmap2)[:,y,x]) / sqrt(C)
// so the pyramid levels are computed as extra small WMMA GEMMs against
// pooled bf16 feature maps instead of re-streaming the 256 MB corr volume.
//
// Pipeline:
//   1) convert fmap1/fmap2 -> (pixel, C) bf16   (A, Bt0)
//   2) pool fmap2 spatially -> Bt1/Bt2/Bt3 bf16 (tiny)
//   3) 4x WMMA bf16 GEMM: L_i = A * Bt_i^T / 16   (N = 8192/2048/512/128)
//   4) bilinear 7x7 lookup per level
// ---------------------------------------------------------------------------

#define HW      8192      // H1*W1 == H2*W2
#define C_DIM   256
#define H1      64
#define W1      128
#define KWIN    7
#define LEVELS  4

typedef __bf16 bf16_t;
typedef bf16_t bf16x8  __attribute__((ext_vector_type(8)));
typedef bf16_t bf16x16 __attribute__((ext_vector_type(16)));
typedef float  f32x8   __attribute__((ext_vector_type(8)));

// ---------------------------------------------------------------------------
// Kernel 1: convert fmap1/fmap2 from (C, HW) fp32 to (HW, C) bf16 row-major.
// c fastest -> contiguous bf16 writes; strided reads absorbed by L2.
// ---------------------------------------------------------------------------
__global__ __launch_bounds__(256)
void convert_kernel(const float* __restrict__ f1, const float* __restrict__ f2,
                    bf16_t* __restrict__ A, bf16_t* __restrict__ Bt0) {
    int idx = blockIdx.x * blockDim.x + threadIdx.x;   // [0, HW*C)
    int c = idx & (C_DIM - 1);
    int p = idx >> 8;
    if (p < HW) {
        A  [p * C_DIM + c] = (bf16_t)f1[(size_t)c * HW + p];
        Bt0[p * C_DIM + c] = (bf16_t)f2[(size_t)c * HW + p];
    }
}

// ---------------------------------------------------------------------------
// Kernel 2: pooled fmap2 pyramid level -> (rowsPad, C) bf16 row-major.
// Row p = y*(W1>>lev) + x; value = mean of fmap2 over the 2^lev x 2^lev block.
// Pad rows in [rows, rowsPad) are zeroed (needed for the 256-wide GEMM tile).
// ---------------------------------------------------------------------------
__global__ __launch_bounds__(256)
void pool_f2_kernel(const float* __restrict__ f2, bf16_t* __restrict__ BtL,
                    int lev, int rowsPad) {
    const int idx = blockIdx.x * blockDim.x + threadIdx.x;  // [0, rowsPad*C)
    const int c = idx & (C_DIM - 1);
    const int p = idx >> 8;
    if (p >= rowsPad) return;

    const int wout = W1 >> lev;
    const int hout = H1 >> lev;
    const int rows = hout * wout;
    if (p >= rows) { BtL[idx] = (bf16_t)0.0f; return; }

    const int x = p % wout;
    const int y = p / wout;
    const int s = 1 << lev;
    const float* src = f2 + (size_t)c * HW + (y * s) * W1 + (x * s);
    float sum = 0.0f;
    for (int dy = 0; dy < s; ++dy)
        for (int dx = 0; dx < s; ++dx)
            sum += src[dy * W1 + dx];
    BtL[p * C_DIM + c] = (bf16_t)(sum / (float)(s * s));
}

// ---------------------------------------------------------------------------
// Kernel 3: corr = A * Bt^T / 16 via v_wmma_f32_16x16x32_bf16.
// Block = 256 threads = 8 waves as 2(M) x 4(N); block tile 64 x 256,
// wave tile 32 x 64 = 2x4 WMMA accumulators.  K loop: 256/32 = 8 steps.
// grid = (colsPad/256, HW/64); stores guarded by n < nCols (L3 padding).
//
// Fragment layout (ISA 7.12.2, 16-bit A 16x32): lane row/col = lane&15,
// kbase = (lane>>4)*8; per lane two contiguous 8-element runs at kbase and
// kbase+16 -> two global_load_b128 per fragment.
// ---------------------------------------------------------------------------
__global__ __launch_bounds__(256)
void corr_gemm_kernel(const bf16_t* __restrict__ A, const bf16_t* __restrict__ Bt,
                      float* __restrict__ corr, int nCols) {
    const int lane  = threadIdx.x & 31;
    const int wave  = threadIdx.x >> 5;
    const int waveM = wave >> 2;            // 0..1
    const int waveN = wave &  3;            // 0..3

    const int mTile = blockIdx.y * 64  + waveM * 32;
    const int nTile = blockIdx.x * 256 + waveN * 64;
    const int l15   = lane & 15;
    const int kb0   = (lane >> 4) << 3;     // 0 or 8

    f32x8 acc[2][4];
#pragma unroll
    for (int tm = 0; tm < 2; ++tm)
#pragma unroll
        for (int tn = 0; tn < 4; ++tn)
#pragma unroll
            for (int r = 0; r < 8; ++r) acc[tm][tn][r] = 0.0f;

#pragma unroll
    for (int k0 = 0; k0 < C_DIM; k0 += 32) {
        const int kb = k0 + kb0;
        bf16x16 afrag[2];
        bf16x16 bfrag[4];
#pragma unroll
        for (int tm = 0; tm < 2; ++tm) {
            const bf16x8* ap =
                (const bf16x8*)(A + (size_t)(mTile + tm * 16 + l15) * C_DIM + kb);
            bf16x8 lo = ap[0];
            bf16x8 hi = ap[2];   // +16 bf16 elements
            afrag[tm] = __builtin_shufflevector(lo, hi,
                0, 1, 2, 3, 4, 5, 6, 7, 8, 9, 10, 11, 12, 13, 14, 15);
        }
#pragma unroll
        for (int tn = 0; tn < 4; ++tn) {
            const bf16x8* bp =
                (const bf16x8*)(Bt + (size_t)(nTile + tn * 16 + l15) * C_DIM + kb);
            bf16x8 lo = bp[0];
            bf16x8 hi = bp[2];
            bfrag[tn] = __builtin_shufflevector(lo, hi,
                0, 1, 2, 3, 4, 5, 6, 7, 8, 9, 10, 11, 12, 13, 14, 15);
        }
#pragma unroll
        for (int tm = 0; tm < 2; ++tm)
#pragma unroll
            for (int tn = 0; tn < 4; ++tn)
                acc[tm][tn] = __builtin_amdgcn_wmma_f32_16x16x32_bf16(
                    false, afrag[tm], false, bfrag[tn],
                    (short)0, acc[tm][tn], false, false);
    }

    // Epilogue: C/D layout — lane n = lane&15; VGPR r -> row r + 8*(lane>>4).
    const int rowOff = (lane >> 4) * 8;
    const float scale = 0.0625f;             // 1/sqrt(256)
#pragma unroll
    for (int tm = 0; tm < 2; ++tm) {
#pragma unroll
        for (int tn = 0; tn < 4; ++tn) {
            const int n = nTile + tn * 16 + l15;
            if (n < nCols) {
#pragma unroll
                for (int r = 0; r < 8; ++r) {
                    const int m = mTile + tm * 16 + rowOff + r;
                    corr[(size_t)m * nCols + n] = acc[tm][tn][r] * scale;
                }
            }
        }
    }
}

// ---------------------------------------------------------------------------
// Kernel 4: bilinear lookup with zero padding.
// Faithful to the reference: x = cg_ch0/2^lev + a (window ROW index),
// y = cg_ch1/2^lev + b (window COL index), offsets 0..6 (no -R shift).
// out[(lev*49 + a*7 + b) * HW + p]
// ---------------------------------------------------------------------------
__device__ __forceinline__
float bilinear_zero(const float* __restrict__ img, int h, int w, float x, float y) {
    const float xf = floorf(x), yf = floorf(y);
    const int x0 = (int)xf, y0 = (int)yf;
    const int x1 = x0 + 1,  y1 = y0 + 1;
    const float wx1 = x - xf, wy1 = y - yf;
    const float wx0 = 1.0f - wx1, wy0 = 1.0f - wy1;
    const bool vx0 = (x0 >= 0) && (x0 < w);
    const bool vx1 = (x1 >= 0) && (x1 < w);
    const bool vy0 = (y0 >= 0) && (y0 < h);
    const bool vy1 = (y1 >= 0) && (y1 < h);
    const int x0c = min(max(x0, 0), w - 1), x1c = min(max(x1, 0), w - 1);
    const int y0c = min(max(y0, 0), h - 1), y1c = min(max(y1, 0), h - 1);
    float v00 = (vx0 && vy0) ? img[y0c * w + x0c] : 0.0f;
    float v10 = (vx1 && vy0) ? img[y0c * w + x1c] : 0.0f;
    float v01 = (vx0 && vy1) ? img[y1c * w + x0c] : 0.0f;
    float v11 = (vx1 && vy1) ? img[y1c * w + x1c] : 0.0f;
    return v00 * wx0 * wy0 + v10 * wx1 * wy0 + v01 * wx0 * wy1 + v11 * wx1 * wy1;
}

__global__ __launch_bounds__(256)
void sample_kernel(const float* __restrict__ cg,
                   const float* __restrict__ l0, const float* __restrict__ l1,
                   const float* __restrict__ l2, const float* __restrict__ l3,
                   float* __restrict__ out) {
    const int idx = blockIdx.x * blockDim.x + threadIdx.x;
    if (idx >= HW * LEVELS * KWIN * KWIN) return;
    const int p  = idx & (HW - 1);          // pixel fastest: coalesced io
    const int ch = idx >> 13;
    const int lev = ch / (KWIN * KWIN);
    const int wnd = ch % (KWIN * KWIN);
    const int a = wnd / KWIN;               // row index of window
    const int b = wnd % KWIN;               // col index of window

    const float inv = 1.0f / (float)(1 << lev);
    const float x = cg[p]      * inv + (float)a;   // ch0 + dy (reference quirk)
    const float y = cg[HW + p] * inv + (float)b;   // ch1 + dx

    const int h = H1 >> lev, w = W1 >> lev;
    const float* img;
    if      (lev == 0) img = l0;
    else if (lev == 1) img = l1;
    else if (lev == 2) img = l2;
    else               img = l3;
    img += (size_t)p * h * w;

    out[idx] = bilinear_zero(img, h, w, x, y);
}

// ---------------------------------------------------------------------------
// Launch
// ---------------------------------------------------------------------------
extern "C" void kernel_launch(void* const* d_in, const int* in_sizes, int n_in,
                              void* d_out, int out_size, void* d_ws, size_t ws_size,
                              hipStream_t stream) {
    (void)in_sizes; (void)n_in; (void)out_size; (void)ws_size;

    const float* fmap1 = (const float*)d_in[0];
    const float* fmap2 = (const float*)d_in[1];
    const float* cgrid = (const float*)d_in[2];
    float* out = (float*)d_out;

    // Workspace layout:
    //   A    bf16 (8192, 256)   4 MB
    //   Bt0  bf16 (8192, 256)   4 MB
    //   Bt1  bf16 (2048, 256)   1 MB
    //   Bt2  bf16 ( 512, 256)   256 KB
    //   Bt3  bf16 ( 256, 256)   128 KB   (rows 128..255 zero pad)
    //   L0   f32  (8192, 8192)  256 MB
    //   L1   f32  (8192, 2048)  64 MB
    //   L2   f32  (8192,  512)  16 MB
    //   L3   f32  (8192,  128)  4 MB
    char* ws = (char*)d_ws;
    bf16_t* A   = (bf16_t*)ws;
    bf16_t* Bt0 = A   + (size_t)HW   * C_DIM;
    bf16_t* Bt1 = Bt0 + (size_t)HW   * C_DIM;
    bf16_t* Bt2 = Bt1 + (size_t)2048 * C_DIM;
    bf16_t* Bt3 = Bt2 + (size_t)512  * C_DIM;
    float*  L0  = (float*)(Bt3 + (size_t)256 * C_DIM);
    float*  L1  = L0 + (size_t)HW * 8192;
    float*  L2  = L1 + (size_t)HW * 2048;
    float*  L3  = L2 + (size_t)HW * 512;

    // 1) convert to bf16 (HW, C)
    convert_kernel<<<(HW * C_DIM) / 256, 256, 0, stream>>>(fmap1, fmap2, A, Bt0);

    // 2) pooled fmap2 pyramids (tiny)
    pool_f2_kernel<<<(2048 * C_DIM) / 256, 256, 0, stream>>>(fmap2, Bt1, 1, 2048);
    pool_f2_kernel<<<( 512 * C_DIM) / 256, 256, 0, stream>>>(fmap2, Bt2, 2,  512);
    pool_f2_kernel<<<( 256 * C_DIM) / 256, 256, 0, stream>>>(fmap2, Bt3, 3,  256);

    // 3) WMMA GEMMs for all pyramid levels: grid (colsPad/256, HW/64)
    corr_gemm_kernel<<<dim3(8192 / 256, HW / 64), 256, 0, stream>>>(A, Bt0, L0, 8192);
    corr_gemm_kernel<<<dim3(2048 / 256, HW / 64), 256, 0, stream>>>(A, Bt1, L1, 2048);
    corr_gemm_kernel<<<dim3( 512 / 256, HW / 64), 256, 0, stream>>>(A, Bt2, L2,  512);
    corr_gemm_kernel<<<dim3(         1, HW / 64), 256, 0, stream>>>(A, Bt3, L3,  128);

    // 4) lookup: HW * 4 * 49 outputs
    const int total = HW * LEVELS * KWIN * KWIN;
    sample_kernel<<<(total + 255) / 256, 256, 0, stream>>>(cgrid, L0, L1, L2, L3, out);
}
